// CEN_GNN_Encoder_76948634075658
// MI455X (gfx1250) — compile-verified, matrix operand order checked by reading, not run on hardware
//
#include <hip/hip_runtime.h>

// ---------------------------------------------------------------------------
// Problem constants (fixed by the reference)
// ---------------------------------------------------------------------------
#define NN   4096
#define EE   32768
#define EF   36864          // EE + NN self loops
#define HH   4

typedef __attribute__((ext_vector_type(16))) __bf16 v16bf;
typedef __attribute__((ext_vector_type(8)))  __bf16 v8bf;
typedef __attribute__((ext_vector_type(8)))  float  v8f;
typedef __attribute__((ext_vector_type(4)))  int    v4i;

typedef __attribute__((address_space(1))) v4i gv4i;   // global (device) int4
typedef __attribute__((address_space(3))) v4i lv4i;   // LDS (shared) int4

#if defined(__HIP_DEVICE_COMPILE__) && __has_builtin(__builtin_amdgcn_global_load_async_to_lds_b128)
#define HAVE_ASYNC_LDS 1
#else
#define HAVE_ASYNC_LDS 0
#endif

// ---------------------------------------------------------------------------
// Small helpers
// ---------------------------------------------------------------------------
__device__ inline unsigned fenc(float f) {
    unsigned u = __float_as_uint(f);
    return (u & 0x80000000u) ? ~u : (u | 0x80000000u);
}
__device__ inline float fdec(unsigned u) {
    unsigned v = (u & 0x80000000u) ? (u & 0x7fffffffu) : ~u;
    return __uint_as_float(v);
}
__device__ inline float wave_reduce(float v) {
    #pragma unroll
    for (int o = 16; o > 0; o >>= 1) v += __shfl_xor(v, o, 32);
    return v;
}

__device__ inline void wait_async0() {
#if HAVE_ASYNC_LDS
#if __has_builtin(__builtin_amdgcn_s_wait_asynccnt)
    __builtin_amdgcn_s_wait_asynccnt(0);
#else
    asm volatile("s_wait_asynccnt 0" ::: "memory");
#endif
#endif
}

// one 16-byte lane-copy global -> LDS
__device__ inline void cp16(const __bf16* g, __bf16* l) {
#if HAVE_ASYNC_LDS
    __builtin_amdgcn_global_load_async_to_lds_b128((gv4i*)g, (lv4i*)l, 0, 0);
#else
    *(v8bf*)l = *(const v8bf*)g;
#endif
}

// ---------------------------------------------------------------------------
// Elementwise / setup kernels
// ---------------------------------------------------------------------------
__global__ void k_zero(float* p, int n) {
    int t = blockIdx.x * blockDim.x + threadIdx.x;
    if (t < n) p[t] = 0.0f;
}
__global__ void k_fill_u32(unsigned* p, unsigned v, int n) {
    int t = blockIdx.x * blockDim.x + threadIdx.x;
    if (t < n) p[t] = v;
}

// concat [x_graph | action | temp_img] -> bf16 x0
__global__ void k_concat(const float* __restrict__ xg, const float* __restrict__ ac,
                         const float* __restrict__ ti, __bf16* __restrict__ xb) {
    int t = blockIdx.x * blockDim.x + threadIdx.x;   // NN*512
    if (t >= NN * 512) return;
    int n = t >> 9, c = t & 511;
    float v;
    if (c < 256)      v = xg[n * 256 + c];
    else if (c < 384) v = ac[n * 128 + (c - 256)];
    else              v = ti[n * 128 + (c - 384)];
    xb[t] = (__bf16)v;
}

// W [K, Nc] f32 row-major -> WT [Nc, K] bf16 row-major (tiled LDS transpose)
__global__ void k_transpose_cvt(const float* __restrict__ W, __bf16* __restrict__ WT,
                                int K, int Nc) {
    __shared__ float tile[32][33];
    int kb = blockIdx.x * 32, nb = blockIdx.y * 32;
    int tx = threadIdx.x, ty = threadIdx.y;          // block (32, 8)
    #pragma unroll
    for (int i = 0; i < 32; i += 8) {
        int k = kb + ty + i, n = nb + tx;
        tile[ty + i][tx] = (k < K && n < Nc) ? W[(size_t)k * Nc + n] : 0.0f;
    }
    __syncthreads();
    #pragma unroll
    for (int i = 0; i < 32; i += 8) {
        int n = nb + ty + i, k = kb + tx;
        if (n < Nc && k < K) WT[(size_t)n * K + k] = (__bf16)tile[tx][ty + i];
    }
}

// ---------------------------------------------------------------------------
// bf16 WMMA GEMM: D[M,Nc] = A[M,K] @ B[K,Nc], Bt = B^T [Nc,K] bf16.
// Block tile 128x256, BK=32, 8 waves as 2x4, each wave 64x64 = 4x4 wmma tiles.
// Double-buffered LDS fed by async global->LDS copies (ASYNCcnt).
// ---------------------------------------------------------------------------
#define BM 128
#define BN 256
#define BK 32
#define LDSTR 40   // padded K-stride in elements (80B, 16B aligned, conflict-free)

__device__ inline v16bf load_frag(const __bf16* base) {
    // elements 0..7 = K[kb+0..kb+7], 8..15 = K[kb+16..kb+23] (ISA bf16 layout)
    v8bf lo = *(const v8bf*)(base);
    v8bf hi = *(const v8bf*)(base + 16);
    v16bf r;
    #pragma unroll
    for (int i = 0; i < 8; i++) { r[i] = lo[i]; r[8 + i] = hi[i]; }
    return r;
}

__global__ __launch_bounds__(256) void k_gemm_bf16(
    const __bf16* __restrict__ A, const __bf16* __restrict__ Bt,
    float* __restrict__ D, const float* __restrict__ bias,
    int M, int Nc, int K, int fuse_bias_act)
{
    __shared__ __bf16 As[2][BM * LDSTR];   // 20 KB
    __shared__ __bf16 Bs[2][BN * LDSTR];   // 40 KB

    int tid  = threadIdx.x;
    int wave = tid >> 5, lane = tid & 31;
    int wm = wave >> 2, wn = wave & 3;                // 2x4 wave grid, 64x64 tiles
    int bm = blockIdx.y * BM, bn = blockIdx.x * BN;

    v8f acc[4][4];
    #pragma unroll
    for (int mt = 0; mt < 4; mt++)
        #pragma unroll
        for (int nt = 0; nt < 4; nt++)
            #pragma unroll
            for (int i = 0; i < 8; i++) acc[mt][nt][i] = 0.0f;

    // staging maps: A tile 128x32 -> thread copies 16 elems (2x b128)
    //               B tile 256x32 -> thread copies its own row (4x b128)
    int arow = tid >> 1, ak = (tid & 1) * 16;
    const __bf16* Ag0 = A  + (size_t)(bm + arow) * K + ak;
    const __bf16* Bg0 = Bt + (size_t)(bn + tid) * K;
    __bf16* Alp[2] = { &As[0][arow * LDSTR + ak], &As[1][arow * LDSTR + ak] };
    __bf16* Blp[2] = { &Bs[0][tid * LDSTR],       &Bs[1][tid * LDSTR] };

    int frow = lane & 15;
    int kb   = (lane >> 4) * 8;

    auto issue = [&](int k0, int buf) {
        const __bf16* Ag = Ag0 + k0;
        const __bf16* Bg = Bg0 + k0;
        __bf16* Al = Alp[buf];
        __bf16* Bl = Blp[buf];
#if HAVE_ASYNC_LDS
        cp16(Ag,      Al);
        cp16(Ag + 8,  Al + 8);
        cp16(Bg,      Bl);
        cp16(Bg + 8,  Bl + 8);
        cp16(Bg + 16, Bl + 16);
        cp16(Bg + 24, Bl + 24);
#else
        // distinct registers first, then stores: keeps loads in flight together
        v8bf a0 = *(const v8bf*)(Ag);
        v8bf a1 = *(const v8bf*)(Ag + 8);
        v8bf b0 = *(const v8bf*)(Bg);
        v8bf b1 = *(const v8bf*)(Bg + 8);
        v8bf b2 = *(const v8bf*)(Bg + 16);
        v8bf b3 = *(const v8bf*)(Bg + 24);
        *(v8bf*)(Al)      = a0;
        *(v8bf*)(Al + 8)  = a1;
        *(v8bf*)(Bl)      = b0;
        *(v8bf*)(Bl + 8)  = b1;
        *(v8bf*)(Bl + 16) = b2;
        *(v8bf*)(Bl + 24) = b3;
#endif
    };

    issue(0, 0);                                     // prologue: tile 0 -> buf 0
    int cur = 0;
    for (int k0 = 0; k0 < K; k0 += BK) {
        wait_async0();                               // own async copies complete
        __syncthreads();                             // all copies visible, prev reads done
        if (k0 + BK < K) issue(k0 + BK, cur ^ 1);    // overlap next tile with compute
        if (k0 + 2 * BK < K) {                       // pre-warm L2 for tile k+2
            __builtin_prefetch(Ag0 + k0 + 2 * BK, 0, 1);
            __builtin_prefetch(Bg0 + k0 + 2 * BK, 0, 1);
        }

        const __bf16* Ab = As[cur];
        const __bf16* Bb = Bs[cur];
        v16bf af[4], bfr[4];
        #pragma unroll
        for (int mt = 0; mt < 4; mt++)
            af[mt] = load_frag(&Ab[(wm * 64 + mt * 16 + frow) * LDSTR + kb]);
        #pragma unroll
        for (int nt = 0; nt < 4; nt++)
            bfr[nt] = load_frag(&Bb[(wn * 64 + nt * 16 + frow) * LDSTR + kb]);

        #pragma unroll
        for (int mt = 0; mt < 4; mt++)
            #pragma unroll
            for (int nt = 0; nt < 4; nt++)
                acc[mt][nt] = __builtin_amdgcn_wmma_f32_16x16x32_bf16(
                    false, af[mt], false, bfr[nt], (short)0, acc[mt][nt], false, false);

        cur ^= 1;
    }

    // store D (C/D layout: VGPR i -> row base+i; lanes 0-15 rows 0-7, 16-31 rows 8-15)
    int rbase = bm + wm * 64;
    int cbase = bn + wn * 64;
    #pragma unroll
    for (int mt = 0; mt < 4; mt++) {
        #pragma unroll
        for (int nt = 0; nt < 4; nt++) {
            int r0 = rbase + mt * 16 + (lane >> 4) * 8;
            int c  = cbase + nt * 16 + (lane & 15);
            #pragma unroll
            for (int i = 0; i < 8; i++) {
                float v = acc[mt][nt][i];
                if (fuse_bias_act) {
                    v += bias[c];
                    v = v > 0.0f ? v : 0.01f * v;
                }
                D[(size_t)(r0 + i) * Nc + c] = v;
            }
        }
    }
}

// ---------------------------------------------------------------------------
// Graph kernels
// ---------------------------------------------------------------------------
__global__ void k_deg_loop(const int* __restrict__ ei, const float* __restrict__ eattr,
                           float* degf, float* lsum) {
    int e = blockIdx.x * blockDim.x + threadIdx.x;
    if (e >= EE) return;
    int dst = ei[EE + e];
    atomicAdd(&degf[dst], 1.0f);
    #pragma unroll
    for (int j = 0; j < 16; j++)
        atomicAdd(&lsum[dst * 16 + j], eattr[e * 16 + j]);
}
__global__ void k_loop_norm(float* lattr, const float* degf) {
    int t = blockIdx.x * blockDim.x + threadIdx.x;   // NN*16
    if (t < NN * 16) lattr[t] /= fmaxf(degf[t >> 4], 1.0f);
}
__global__ void k_build_edges(const int* __restrict__ ei, const float* __restrict__ eattr,
                              const float* __restrict__ lattr,
                              int* srcf, int* dstf, float* eaf) {
    int t = blockIdx.x * blockDim.x + threadIdx.x;
    if (t >= EF) return;
    if (t < EE) {
        srcf[t] = ei[t];
        dstf[t] = ei[EE + t];
        #pragma unroll
        for (int j = 0; j < 16; j++) eaf[t * 16 + j] = eattr[t * 16 + j];
    } else {
        int n = t - EE;
        srcf[t] = n; dstf[t] = n;
        #pragma unroll
        for (int j = 0; j < 16; j++) eaf[t * 16 + j] = lattr[n * 16 + j];
    }
}

// Ae[k,hd] = sum_c We[k, hd*C+c] * a_e[hd, c]    (16 x 4, tiny)
__global__ void k_ae(const float* __restrict__ We, const float* __restrict__ a_e,
                     float* Ae, int C) {
    int t = threadIdx.x;                             // 64 threads
    if (t >= 16 * HH) return;
    int k = t >> 2, hd = t & 3;
    const float* w = We + (size_t)k * (HH * C) + (size_t)hd * C;
    const float* a = a_e + (size_t)hd * C;
    float s = 0.0f;
    for (int c = 0; c < C; c++) s += w[c] * a[c];
    Ae[t] = s;
}

// asrc[n,hd] = h[n,hd,:]·a_s ; adst likewise (one wave per (n,hd))
__global__ __launch_bounds__(256) void k_alpha(const float* __restrict__ h,
                                               const float* __restrict__ a_s,
                                               const float* __restrict__ a_d,
                                               float* asrc, float* adst, int C) {
    int wave = threadIdx.x >> 5, lane = threadIdx.x & 31;
    int idx = blockIdx.x * 8 + wave;                 // (n, hd)
    if (idx >= NN * HH) return;
    int n = idx >> 2, hd = idx & 3;
    const float* hr = h + (size_t)n * (HH * C) + (size_t)hd * C;
    const float* as = a_s + (size_t)hd * C;
    const float* ad = a_d + (size_t)hd * C;
    float s0 = 0.0f, s1 = 0.0f;
    for (int c = lane; c < C; c += 32) {
        float hv = hr[c];
        s0 += hv * as[c];
        s1 += hv * ad[c];
    }
    s0 = wave_reduce(s0);
    s1 = wave_reduce(s1);
    if (lane == 0) { asrc[idx] = s0; adst[idx] = s1; }
}

// e = leaky(asrc[src]+adst[dst]+eattr@Ae, 0.2); atomicMax per (dst, head)
__global__ void k_elogits(const float* __restrict__ eaf, const int* __restrict__ srcf,
                          const int* __restrict__ dstf, const float* __restrict__ Ae,
                          const float* __restrict__ asrc, const float* __restrict__ adst,
                          float* elog, unsigned* menc) {
    int f = blockIdx.x * blockDim.x + threadIdx.x;
    if (f >= EF) return;
    float ea[16];
    #pragma unroll
    for (int j = 0; j < 16; j++) ea[j] = eaf[f * 16 + j];
    int sf = srcf[f], df = dstf[f];
    #pragma unroll
    for (int hd = 0; hd < HH; hd++) {
        float s = asrc[sf * HH + hd] + adst[df * HH + hd];
        #pragma unroll
        for (int k = 0; k < 16; k++) s += ea[k] * Ae[k * HH + hd];
        s = s > 0.0f ? s : 0.2f * s;
        elog[f * HH + hd] = s;
        atomicMax(&menc[df * HH + hd], fenc(s));
    }
}

__global__ void k_expsum(const float* __restrict__ elog, const int* __restrict__ dstf,
                         const unsigned* __restrict__ menc, float* exs, float* ssum) {
    int t = blockIdx.x * blockDim.x + threadIdx.x;   // EF*HH
    if (t >= EF * HH) return;
    int f = t >> 2, hd = t & 3;
    int df = dstf[f];
    float m = fdec(menc[df * HH + hd]);
    float e = __expf(elog[t] - m);
    exs[t] = e;
    atomicAdd(&ssum[df * HH + hd], e);
}

__global__ void k_out_init(float* out, const float* __restrict__ bias, int HC) {
    int t = blockIdx.x * blockDim.x + threadIdx.x;   // NN*HC
    if (t < NN * HC) out[t] = bias[t & (HC - 1)];    // HC is a power of two
}

// out[dst, j] += (exs/ssum) * h[src, j]  — float4 vectorized atomic scatter
__global__ __launch_bounds__(256) void k_aggregate(
    const float* __restrict__ h, const int* __restrict__ srcf,
    const int* __restrict__ dstf, const float* __restrict__ exs,
    const float* __restrict__ ssum, float* out, int C) {
    int HC = HH * C;
    int perE = HC >> 2;
    int t = blockIdx.x * blockDim.x + threadIdx.x;
    if (t >= EF * perE) return;
    int f  = t / perE;
    int j4 = (t - f * perE) << 2;
    int hd = j4 / C;
    int sf = srcf[f], df = dstf[f];
    float alpha = exs[f * HH + hd] / ssum[df * HH + hd];
    const float4 hv = *(const float4*)(h + (size_t)sf * HC + j4);
    float* o = out + (size_t)df * HC + j4;
    atomicAdd(o + 0, alpha * hv.x);
    atomicAdd(o + 1, alpha * hv.y);
    atomicAdd(o + 2, alpha * hv.z);
    atomicAdd(o + 3, alpha * hv.w);
}

__global__ void k_act_cvt(float* x, __bf16* xb, int n) {
    int t = blockIdx.x * blockDim.x + threadIdx.x;
    if (t >= n) return;
    float v = x[t];
    v = v > 0.0f ? v : 0.01f * v;
    x[t]  = v;
    xb[t] = (__bf16)v;
}

// ---------------------------------------------------------------------------
// Pooling
// ---------------------------------------------------------------------------
__global__ void k_pool_acc(const float* __restrict__ hfc, const int* __restrict__ batch,
                           float* psum, float* pcnt) {
    int t = blockIdx.x * blockDim.x + threadIdx.x;   // NN*256
    if (t >= NN * 256) return;
    int n = t >> 8, c = t & 255;
    int g = batch[n];
    atomicAdd(&psum[g * 256 + c], hfc[t]);
    if (c == 0) atomicAdd(&pcnt[g], 1.0f);
}
__global__ void k_pool_final(const float* psum, const float* pcnt, float* out) {
    int t = blockIdx.x * blockDim.x + threadIdx.x;   // 8*256
    if (t < 8 * 256) out[t] = psum[t] / fmaxf(pcnt[t >> 8], 1.0f);
}

// ---------------------------------------------------------------------------
// Host-side layer driver
// ---------------------------------------------------------------------------
static void gat_layer(const __bf16* xb, int Kin,
                      const float* W, __bf16* Wt,
                      const float* We, const float* a_s, const float* a_d,
                      const float* a_e, const float* bias, int C,
                      float* h, float* outb, __bf16* xnextb,
                      const int* srcf, const int* dstf, const float* eaf,
                      float* Ae, float* asrc, float* adst, float* elog, float* exs,
                      unsigned* menc, float* ssum, hipStream_t stream) {
    int HC = HH * C;
    dim3 tb(32, 8);
    k_transpose_cvt<<<dim3(Kin / 32, HC / 32), tb, 0, stream>>>(W, Wt, Kin, HC);
    k_gemm_bf16<<<dim3(HC / BN, NN / BM), 256, 0, stream>>>(
        xb, Wt, h, nullptr, NN, HC, Kin, 0);
    k_ae<<<1, 64, 0, stream>>>(We, a_e, Ae, C);
    k_alpha<<<(NN * HH) / 8, 256, 0, stream>>>(h, a_s, a_d, asrc, adst, C);
    k_fill_u32<<<(NN * HH + 255) / 256, 256, 0, stream>>>(menc, 0x007FFFFFu, NN * HH);
    k_zero<<<(NN * HH + 255) / 256, 256, 0, stream>>>(ssum, NN * HH);
    k_elogits<<<(EF + 255) / 256, 256, 0, stream>>>(eaf, srcf, dstf, Ae, asrc, adst, elog, menc);
    k_expsum<<<(EF * HH + 255) / 256, 256, 0, stream>>>(elog, dstf, menc, exs, ssum);
    int nOut = NN * HC;
    k_out_init<<<(nOut + 255) / 256, 256, 0, stream>>>(outb, bias, HC);
    int perE = HC >> 2;
    long long tot = (long long)EF * perE;
    k_aggregate<<<(unsigned)((tot + 255) / 256), 256, 0, stream>>>(
        h, srcf, dstf, exs, ssum, outb, C);
    k_act_cvt<<<(nOut + 255) / 256, 256, 0, stream>>>(outb, xnextb, nOut);
}

// ---------------------------------------------------------------------------
// Entry point
// ---------------------------------------------------------------------------
extern "C" void kernel_launch(void* const* d_in, const int* in_sizes, int n_in,
                              void* d_out, int out_size, void* d_ws, size_t ws_size,
                              hipStream_t stream) {
    const float* x_graph   = (const float*)d_in[0];
    const float* action    = (const float*)d_in[1];
    const float* temp_img  = (const float*)d_in[2];
    const float* edge_attr = (const float*)d_in[3];
    const int*   edge_index= (const int*)d_in[4];
    const int*   batch     = (const int*)d_in[5];
    const float* W1  = (const float*)d_in[6];
    const float* as1 = (const float*)d_in[7];
    const float* ad1 = (const float*)d_in[8];
    const float* ae1 = (const float*)d_in[9];
    const float* We1 = (const float*)d_in[10];
    const float* b1  = (const float*)d_in[11];
    const float* W2  = (const float*)d_in[12];
    const float* as2 = (const float*)d_in[13];
    const float* ad2 = (const float*)d_in[14];
    const float* ae2 = (const float*)d_in[15];
    const float* We2 = (const float*)d_in[16];
    const float* b2  = (const float*)d_in[17];
    const float* Wout= (const float*)d_in[18];
    const float* bout= (const float*)d_in[19];

    char* ws = (char*)d_ws;
    size_t off = 0;
    auto alloc = [&](size_t bytes) -> char* {
        off = (off + 255) & ~(size_t)255;
        char* p = ws + off;
        off += bytes;
        return p;
    };

    __bf16*  x0b  = (__bf16*)alloc((size_t)NN * 512 * 2);
    __bf16*  w1t  = (__bf16*)alloc((size_t)2048 * 512 * 2);
    float*   h1   = (float*) alloc((size_t)NN * 2048 * 4);
    float*   out1 = (float*) alloc((size_t)NN * 2048 * 4);
    __bf16*  x1b  = (__bf16*)alloc((size_t)NN * 2048 * 2);
    __bf16*  w2t  = (__bf16*)alloc((size_t)8192 * 2048 * 2);
    float*   h2   = (float*) alloc((size_t)NN * 8192 * 4);
    float*   out2 = (float*) alloc((size_t)NN * 8192 * 4);
    __bf16*  x2b  = (__bf16*)alloc((size_t)NN * 8192 * 2);
    __bf16*  wot  = (__bf16*)alloc((size_t)256 * 8192 * 2);
    float*   hfc  = (float*) alloc((size_t)NN * 256 * 4);
    float*   degf = (float*) alloc((size_t)NN * 4);
    float*   lattr= (float*) alloc((size_t)NN * 16 * 4);
    int*     srcf = (int*)   alloc((size_t)EF * 4);
    int*     dstf = (int*)   alloc((size_t)EF * 4);
    float*   eaf  = (float*) alloc((size_t)EF * 16 * 4);
    float*   Ae   = (float*) alloc(16 * HH * 4);
    float*   asrc = (float*) alloc((size_t)NN * HH * 4);
    float*   adst = (float*) alloc((size_t)NN * HH * 4);
    float*   elog = (float*) alloc((size_t)EF * HH * 4);
    float*   exs  = (float*) alloc((size_t)EF * HH * 4);
    unsigned* menc= (unsigned*)alloc((size_t)NN * HH * 4);
    float*   ssum = (float*) alloc((size_t)NN * HH * 4);
    float*   psum = (float*) alloc(8 * 256 * 4);
    float*   pcnt = (float*) alloc(8 * 4);
    (void)ws_size; (void)in_sizes; (void)n_in; (void)out_size;

    // 1. concat inputs -> bf16 node features
    k_concat<<<(NN * 512) / 256, 256, 0, stream>>>(x_graph, action, temp_img, x0b);

    // 2. self-loop edge attrs (mean of incoming) + concatenated edge lists
    k_zero<<<(NN + 255) / 256, 256, 0, stream>>>(degf, NN);
    k_zero<<<(NN * 16 + 255) / 256, 256, 0, stream>>>(lattr, NN * 16);
    k_deg_loop<<<(EE + 255) / 256, 256, 0, stream>>>(edge_index, edge_attr, degf, lattr);
    k_loop_norm<<<(NN * 16 + 255) / 256, 256, 0, stream>>>(lattr, degf);
    k_build_edges<<<(EF + 255) / 256, 256, 0, stream>>>(edge_index, edge_attr, lattr,
                                                        srcf, dstf, eaf);

    // 3. GAT layer 1 (512 -> 4x512) and layer 2 (2048 -> 4x2048)
    gat_layer(x0b, 512, W1, w1t, We1, as1, ad1, ae1, b1, 512,
              h1, out1, x1b, srcf, dstf, eaf, Ae, asrc, adst, elog, exs,
              menc, ssum, stream);
    gat_layer(x1b, 2048, W2, w2t, We2, as2, ad2, ae2, b2, 2048,
              h2, out2, x2b, srcf, dstf, eaf, Ae, asrc, adst, elog, exs,
              menc, ssum, stream);

    // 4. fc_output (8192 -> 256) with fused bias + leaky_relu
    dim3 tb(32, 8);
    k_transpose_cvt<<<dim3(8192 / 32, 256 / 32), tb, 0, stream>>>(Wout, wot, 8192, 256);
    k_gemm_bf16<<<dim3(256 / BN, NN / BM), 256, 0, stream>>>(
        x2b, wot, hfc, bout, NN, 256, 8192, 1);

    // 5. mean pool per graph
    k_zero<<<(8 * 256 + 255) / 256, 256, 0, stream>>>(psum, 8 * 256);
    k_zero<<<1, 32, 0, stream>>>(pcnt, 8);
    k_pool_acc<<<(NN * 256) / 256, 256, 0, stream>>>(hfc, batch, psum, pcnt);
    k_pool_final<<<(8 * 256 + 255) / 256, 256, 0, stream>>>(psum, pcnt, (float*)d_out);
}
